// MP_PDE_Solver_2D_40510131536547
// MI455X (gfx1250) — matrix operand order repaired and checked by reading, not compiled, not run on hardware
//
#include <hip/hip_runtime.h>
#include <hip/hip_bf16.h>

// ---------------------------------------------------------------------------
// MP-PDE GNN solver for gfx1250 (MI455X).
//  - All GEMMs on matrix cores via v_wmma_f32_16x16x32_f16.
//  - 64 rows (4 M-tiles) per workgroup: each B fragment loaded once per wave
//    and reused across 4 A fragments -> ~4x FLOP per L2 byte of weights.
//  - h kept as an f16 mirror; gathers load packed half-pairs (b32) straight
//    into the WMMA A-fragment layout in LDS.
// ---------------------------------------------------------------------------

#define NN    50000
#define EE    600000
#define HH    128
#define LL    6
#define K_MSG 260    // 2H + 1 + 2 + 1
#define K_UPD 257    // 2H + 1
#define KP    288    // padded K (9 chunks of 32)
#define EPSF  1e-5f
#define DTOUT 0.001f // DT * 0.1

typedef __attribute__((ext_vector_type(16))) _Float16 v16h;
typedef __attribute__((ext_vector_type(8)))  float    v8f;

union HalfPair { _Float16 h[2]; unsigned w; };

// ---- A-fragment half-index for 16-bit 16x32 WMMA operand -------------------
__device__ __forceinline__ int afrag_idx(int m, int kk /*0..31*/) {
    int hi = (kk >> 3) & 1;
    int j  = (kk & 7) | ((kk & 16) ? 8 : 0);
    return (m + hi * 16) * 16 + j;
}

__device__ __forceinline__ v8f wmma_f16(v16h a, v16h b, v8f c) {
    return __builtin_amdgcn_wmma_f32_16x16x32_f16(false, a, false, b,
                                                  (short)0, c, false, false);
}

// ---------------------------------------------------------------------------
// Pack fp32 weights [Lc][K][128] into WMMA B-fragment layout (f16, K padded).
// ---------------------------------------------------------------------------
__global__ void k_pack(const float* __restrict__ src, _Float16* __restrict__ dst,
                       int Lc, int K, int KC) {
    long total = (long)Lc * KC * 8 * 512;
    for (long i = blockIdx.x * (long)blockDim.x + threadIdx.x; i < total;
         i += (long)gridDim.x * blockDim.x) {
        int j    = (int)(i & 15);
        int lane = (int)((i >> 4) & 31);
        int nt   = (int)((i >> 9) & 7);
        long r   = i >> 12;
        int kc   = (int)(r % KC);
        int l    = (int)(r / KC);
        int k    = kc * 32 + ((lane >> 4) & 1) * 8 + (j & 7) + ((j >> 3) & 1) * 16;
        int n    = nt * 16 + (lane & 15);
        _Float16 v = (_Float16)0.f;
        if (k < K) v = (_Float16)src[((long)l * K + k) * HH + n];
        dst[i] = v;
    }
}

// ---------------------------------------------------------------------------
__global__ void k_deg(const int* __restrict__ ei, float* __restrict__ deg, int E_) {
    int e = blockIdx.x * blockDim.x + threadIdx.x;
    if (e < E_) atomicAdd(&deg[ei[E_ + e]], 1.0f);
}

// ---------------------------------------------------------------------------
// Embedding layer 1: [u, px, py, var] @ W(4x128) + b ; accumulate BN sums.
// ---------------------------------------------------------------------------
__global__ void k_emb1(const float* __restrict__ u, const float* __restrict__ pos,
                       const float* __restrict__ W, const float* __restrict__ b,
                       float* __restrict__ pre, float* __restrict__ S1,
                       float* __restrict__ S2, int N_) {
    int c      = threadIdx.x & 127;
    int group  = (blockIdx.x * blockDim.x + threadIdx.x) >> 7;
    int groups = (gridDim.x * blockDim.x) >> 7;
    float w0 = W[c], w1 = W[HH + c], w2 = W[2 * HH + c], w3 = W[3 * HH + c];
    float bb = b[c];
    float s1 = 0.f, s2 = 0.f;
    for (int n = group; n < N_; n += groups) {
        float v = u[n] * w0 + pos[n * 3 + 1] * w1 + pos[n * 3 + 2] * w2 +
                  pos[n * 3 + 0] * w3 + bb;
        pre[(long)n * HH + c] = v;
        s1 += v; s2 += v * v;
    }
    atomicAdd(&S1[c], s1);
    atomicAdd(&S2[c], s2);
}

// ---------------------------------------------------------------------------
// Batch-norm apply; writes f32 result and f16 mirror (WMMA A operand source).
// ---------------------------------------------------------------------------
__global__ void k_bn_apply(const float* __restrict__ pre, float* __restrict__ dst,
                           _Float16* __restrict__ dst16,
                           const float* __restrict__ g, const float* __restrict__ b,
                           const float* __restrict__ S1, const float* __restrict__ S2,
                           int relu, int N_) {
    int i = blockIdx.x * blockDim.x + threadIdx.x;
    if (i >= N_ * HH) return;
    int c = i & 127;
    float invN = 1.0f / (float)N_;
    float mu   = S1[c] * invN;
    float var  = S2[c] * invN - mu * mu;
    float v = g[c] * (pre[i] - mu) * rsqrtf(var + EPSF) + b[c];
    if (relu) v = fmaxf(v, 0.f);
    dst[i]   = v;
    dst16[i] = (_Float16)v;
}

// ---------------------------------------------------------------------------
// Dense node GEMM (embedding layer 2): X(f16 mirror) @ W + b -> pre + BN sums.
// 64 nodes / block, 4 M-tiles, 4 waves x 2 N-tiles, B reused across M-tiles.
// ---------------------------------------------------------------------------
__global__ __launch_bounds__(128) void k_gemm64(
    const _Float16* __restrict__ Xf, const _Float16* __restrict__ Wp,
    const float* __restrict__ bias, float* __restrict__ pre,
    float* __restrict__ S1, float* __restrict__ S2, int N_) {
    __shared__ __align__(32) _Float16 sA[4 * 4 * 512];
    int tid = threadIdx.x;
    int n0t = blockIdx.x * 64;
    for (int idx = tid; idx < 64 * 64; idx += 128) {
        int m = idx >> 6;
        int k = (idx & 63) * 2;
        int n = n0t + m; if (n >= N_) n = N_ - 1;
        unsigned v = *(const unsigned*)(Xf + (long)n * HH + k);
        int mt = m >> 4, mm = m & 15;
        *(unsigned*)(sA + (mt * 4 + (k >> 5)) * 512 + afrag_idx(mm, k & 31)) = v;
    }
    __syncthreads();
    int lane = tid & 31, wave = tid >> 5;
    for (int t = 0; t < 2; ++t) {
        int nt = wave * 2 + t;
        v8f acc[4] = {{}, {}, {}, {}};
#pragma unroll
        for (int kc = 0; kc < 4; ++kc) {
            v16h b = *(const v16h*)(Wp + ((kc * 8 + nt) * 512 + lane * 16));
#pragma unroll
            for (int mt = 0; mt < 4; ++mt) {
                v16h a = *(const v16h*)(sA + (mt * 4 + kc) * 512 + lane * 16);
                acc[mt] = wmma_f16(a, b, acc[mt]);
            }
        }
        int ch = nt * 16 + (lane & 15);
        float bv = bias[ch];
        int mb = (lane >= 16) ? 8 : 0;
        float s1 = 0.f, s2 = 0.f;
#pragma unroll
        for (int mt = 0; mt < 4; ++mt) {
#pragma unroll
            for (int r = 0; r < 8; ++r) {
                int n = n0t + mt * 16 + mb + r;
                if (n < N_) {
                    float v = acc[mt][r] + bv;
                    pre[(long)n * HH + ch] = v;
                    s1 += v; s2 += v * v;
                }
            }
        }
        atomicAdd(&S1[ch], s1);
        atomicAdd(&S2[ch], s2);
    }
}

// ---------------------------------------------------------------------------
// Edge message kernel: 64 edges / block (4 M-tiles).
//   msg = relu(relu([h_d|h_s|du|dpx|dpy|var] @ W1 + b1) @ W2 + b2)
//   agg[dst] += msg   (f32 atomics, L2-resident)
// ---------------------------------------------------------------------------
__global__ __launch_bounds__(128) void k_edge(
    const _Float16* __restrict__ hf, const float* __restrict__ u,
    const float* __restrict__ pos, const int* __restrict__ ei,
    const _Float16* __restrict__ W1p, const float* __restrict__ b1,
    const _Float16* __restrict__ W2p, const float* __restrict__ b2,
    float* __restrict__ agg, int E_) {
    __shared__ __align__(32) _Float16 sA[4 * 9 * 512];  // 36.9 KB msg_in frags
    __shared__ __align__(32) _Float16 sM[4 * 4 * 512];  // 16.4 KB hidden frags
    __shared__ int   sSrc[64], sDst[64];
    __shared__ float sVals[64][4];

    int tid = threadIdx.x;
    int e0  = blockIdx.x * 64;
    __builtin_prefetch(W1p, 0, 2);
    __builtin_prefetch(W2p, 0, 2);
    if (tid < 64) {
        int e = e0 + tid;
        int s = 0, d = 0;
        if (e < E_) { s = ei[e]; d = ei[E_ + e]; }
        sSrc[tid] = s; sDst[tid] = d;
        sVals[tid][0] = u[d] - u[s];
        sVals[tid][1] = pos[d * 3 + 1] - pos[s * 3 + 1];
        sVals[tid][2] = pos[d * 3 + 2] - pos[s * 3 + 2];
        sVals[tid][3] = pos[d * 3 + 0];
    }
    __syncthreads();
    // gather msg_in as packed half-pairs into A-fragment layout
    for (int idx = tid; idx < 64 * (KP / 2); idx += 128) {
        int m = idx / (KP / 2);
        int k = (idx % (KP / 2)) * 2;
        unsigned v = 0u;
        if (k < 2 * HH) {
            int node = (k < HH) ? sDst[m] : sSrc[m];
            v = *(const unsigned*)(hf + (long)node * HH + (k & 127));
        } else if (k < K_MSG) {
            HalfPair p;
            p.h[0] = (_Float16)sVals[m][k - 2 * HH];
            p.h[1] = (_Float16)sVals[m][k - 2 * HH + 1];
            v = p.w;
        }
        int mt = m >> 4, mm = m & 15;
        *(unsigned*)(sA + (mt * 9 + (k >> 5)) * 512 + afrag_idx(mm, k & 31)) = v;
    }
    __syncthreads();

    int lane = tid & 31, wave = tid >> 5;
    for (int t = 0; t < 2; ++t) {
        int nt = wave * 2 + t;
        v8f acc[4] = {{}, {}, {}, {}};
#pragma unroll
        for (int kc = 0; kc < 9; ++kc) {
            v16h b = *(const v16h*)(W1p + ((kc * 8 + nt) * 512 + lane * 16));
#pragma unroll
            for (int mt = 0; mt < 4; ++mt) {
                v16h a = *(const v16h*)(sA + (mt * 9 + kc) * 512 + lane * 16);
                acc[mt] = wmma_f16(a, b, acc[mt]);
            }
        }
        int ch = nt * 16 + (lane & 15);
        float bv = b1[ch];
        int mb  = (lane >= 16) ? 8 : 0;
        int kc2 = ch >> 5, kk = ch & 31;
#pragma unroll
        for (int mt = 0; mt < 4; ++mt) {
#pragma unroll
            for (int r = 0; r < 8; ++r) {
                float v = fmaxf(acc[mt][r] + bv, 0.f);
                sM[(mt * 4 + kc2) * 512 + afrag_idx(mb + r, kk)] = (_Float16)v;
            }
        }
    }
    __syncthreads();
    for (int t = 0; t < 2; ++t) {
        int nt = wave * 2 + t;
        v8f acc[4] = {{}, {}, {}, {}};
#pragma unroll
        for (int kc = 0; kc < 4; ++kc) {
            v16h b = *(const v16h*)(W2p + ((kc * 8 + nt) * 512 + lane * 16));
#pragma unroll
            for (int mt = 0; mt < 4; ++mt) {
                v16h a = *(const v16h*)(sM + (mt * 4 + kc) * 512 + lane * 16);
                acc[mt] = wmma_f16(a, b, acc[mt]);
            }
        }
        int ch = nt * 16 + (lane & 15);
        float bv = b2[ch];
        int mb = (lane >= 16) ? 8 : 0;
#pragma unroll
        for (int mt = 0; mt < 4; ++mt) {
#pragma unroll
            for (int r = 0; r < 8; ++r) {
                int m = mt * 16 + mb + r;
                if (e0 + m < E_) {
                    float v = fmaxf(acc[mt][r] + bv, 0.f);
                    atomicAdd(&agg[(long)sDst[m] * HH + ch], v);
                }
            }
        }
    }
}

// ---------------------------------------------------------------------------
// Node update kernel: up = relu(relu([h|agg/deg|var]@U1+b1)@U2+b2);
// pre = h + up; accumulate BN sums.  64 nodes / block (4 M-tiles).
// ---------------------------------------------------------------------------
__global__ __launch_bounds__(128) void k_update(
    const _Float16* __restrict__ hf, const float* __restrict__ h,
    const float* __restrict__ agg, const float* __restrict__ deg,
    const float* __restrict__ pos,
    const _Float16* __restrict__ W1p, const float* __restrict__ b1,
    const _Float16* __restrict__ W2p, const float* __restrict__ b2,
    float* __restrict__ pre, float* __restrict__ S1, float* __restrict__ S2,
    int N_) {
    __shared__ __align__(32) _Float16 sA[4 * 9 * 512];
    __shared__ __align__(32) _Float16 sM[4 * 4 * 512];
    __shared__ float sInvDeg[64];
    __shared__ float sVar[64];

    int tid = threadIdx.x;
    int n0t = blockIdx.x * 64;
    __builtin_prefetch(W1p, 0, 2);
    __builtin_prefetch(W2p, 0, 2);
    if (tid < 64) {
        int n = n0t + tid; if (n >= N_) n = N_ - 1;
        sInvDeg[tid] = 1.0f / fmaxf(deg[n], 1.0f);
        sVar[tid]    = pos[n * 3 + 0];
    }
    __syncthreads();
    for (int idx = tid; idx < 64 * (KP / 2); idx += 128) {
        int m = idx / (KP / 2);
        int k = (idx % (KP / 2)) * 2;
        int n = n0t + m; if (n >= N_) n = N_ - 1;
        unsigned v = 0u;
        if (k < HH) {
            v = *(const unsigned*)(hf + (long)n * HH + k);
        } else if (k < 2 * HH) {
            float2 a2 = *(const float2*)(agg + (long)n * HH + (k - HH));
            HalfPair p;
            p.h[0] = (_Float16)(a2.x * sInvDeg[m]);
            p.h[1] = (_Float16)(a2.y * sInvDeg[m]);
            v = p.w;
        } else if (k == 2 * HH) {
            HalfPair p;
            p.h[0] = (_Float16)sVar[m];
            p.h[1] = (_Float16)0.f;
            v = p.w;
        }
        int mt = m >> 4, mm = m & 15;
        *(unsigned*)(sA + (mt * 9 + (k >> 5)) * 512 + afrag_idx(mm, k & 31)) = v;
    }
    __syncthreads();

    int lane = tid & 31, wave = tid >> 5;
    for (int t = 0; t < 2; ++t) {
        int nt = wave * 2 + t;
        v8f acc[4] = {{}, {}, {}, {}};
#pragma unroll
        for (int kc = 0; kc < 9; ++kc) {
            v16h b = *(const v16h*)(W1p + ((kc * 8 + nt) * 512 + lane * 16));
#pragma unroll
            for (int mt = 0; mt < 4; ++mt) {
                v16h a = *(const v16h*)(sA + (mt * 9 + kc) * 512 + lane * 16);
                acc[mt] = wmma_f16(a, b, acc[mt]);
            }
        }
        int ch = nt * 16 + (lane & 15);
        float bv = b1[ch];
        int mb  = (lane >= 16) ? 8 : 0;
        int kc2 = ch >> 5, kk = ch & 31;
#pragma unroll
        for (int mt = 0; mt < 4; ++mt) {
#pragma unroll
            for (int r = 0; r < 8; ++r) {
                float v = fmaxf(acc[mt][r] + bv, 0.f);
                sM[(mt * 4 + kc2) * 512 + afrag_idx(mb + r, kk)] = (_Float16)v;
            }
        }
    }
    __syncthreads();
    for (int t = 0; t < 2; ++t) {
        int nt = wave * 2 + t;
        v8f acc[4] = {{}, {}, {}, {}};
#pragma unroll
        for (int kc = 0; kc < 4; ++kc) {
            v16h b = *(const v16h*)(W2p + ((kc * 8 + nt) * 512 + lane * 16));
#pragma unroll
            for (int mt = 0; mt < 4; ++mt) {
                v16h a = *(const v16h*)(sM + (mt * 4 + kc) * 512 + lane * 16);
                acc[mt] = wmma_f16(a, b, acc[mt]);
            }
        }
        int ch = nt * 16 + (lane & 15);
        float bv = b2[ch];
        int mb = (lane >= 16) ? 8 : 0;
        float s1 = 0.f, s2 = 0.f;
#pragma unroll
        for (int mt = 0; mt < 4; ++mt) {
#pragma unroll
            for (int r = 0; r < 8; ++r) {
                int n = n0t + mt * 16 + mb + r;
                if (n < N_) {
                    float v = fmaxf(acc[mt][r] + bv, 0.f) + h[(long)n * HH + ch];
                    pre[(long)n * HH + ch] = v;
                    s1 += v; s2 += v * v;
                }
            }
        }
        atomicAdd(&S1[ch], s1);
        atomicAdd(&S2[ch], s2);
    }
}

// ---------------------------------------------------------------------------
// Conv head: lazily re-evaluated conv1->conv2->conv3 chain per node.
// ---------------------------------------------------------------------------
__global__ __launch_bounds__(64) void k_head(
    const float* __restrict__ h,
    const float* __restrict__ c1W, const float* __restrict__ c1b,
    const float* __restrict__ c2W, const float* __restrict__ c2b,
    const float* __restrict__ c3W, const float* __restrict__ c3b,
    float* __restrict__ out, int N_) {
    __shared__ float sh[64 * 132];
    int tid = threadIdx.x;
    int n0  = blockIdx.x * 64;
    for (int idx = tid; idx < 64 * HH; idx += 64) {
        int m = idx >> 7, k = idx & 127;
        int n = n0 + m;
        sh[m * 132 + k] = (n < N_) ? h[(long)n * HH + k] : 0.f;
    }
    __syncthreads();
    if (n0 + tid >= N_) return;
    const float* row = &sh[tid * 132];
    float acc3 = c3b[0];
    for (int ic2 = 0; ic2 < 8; ++ic2) {
        for (int k2 = 0; k2 < 8; ++k2) {
            float a2 = c2b[ic2];
            for (int ic1 = 0; ic1 < 4; ++ic1) {
#pragma unroll
                for (int k1 = 0; k1 < 12; ++k1) {
                    int t1 = k2 * 3 + k1;
                    float a1 = c1b[ic1];
#pragma unroll
                    for (int k0 = 0; k0 < 16; ++k0)
                        a1 += row[t1 * 3 + k0] * c1W[ic1 * 16 + k0];
                    a1 = fmaxf(a1, 0.f);
                    a2 += a1 * c2W[(ic2 * 4 + ic1) * 12 + k1];
                }
            }
            a2 = fmaxf(a2, 0.f);
            acc3 += a2 * c3W[ic2 * 8 + k2];
        }
    }
    out[n0 + tid] = acc3 * DTOUT;
}

// ---------------------------------------------------------------------------
extern "C" void kernel_launch(void* const* d_in, const int* in_sizes, int n_in,
                              void* d_out, int out_size, void* d_ws, size_t ws_size,
                              hipStream_t stream) {
    const float* u       = (const float*)d_in[0];
    const float* pos     = (const float*)d_in[1];
    const int*   ei      = (const int*)  d_in[2];
    const float* emb_W1  = (const float*)d_in[3];
    const float* emb_b1  = (const float*)d_in[4];
    const float* emb_g1  = (const float*)d_in[5];
    const float* emb_be1 = (const float*)d_in[6];
    const float* emb_W2  = (const float*)d_in[7];
    const float* emb_b2  = (const float*)d_in[8];
    const float* emb_g2  = (const float*)d_in[9];
    const float* emb_be2 = (const float*)d_in[10];
    const float* m1_W    = (const float*)d_in[11];
    const float* m1_b    = (const float*)d_in[12];
    const float* m2_W    = (const float*)d_in[13];
    const float* m2_b    = (const float*)d_in[14];
    const float* u1_W    = (const float*)d_in[15];
    const float* u1_b    = (const float*)d_in[16];
    const float* u2_W    = (const float*)d_in[17];
    const float* u2_b    = (const float*)d_in[18];
    const float* bn_g    = (const float*)d_in[19];
    const float* bn_b    = (const float*)d_in[20];
    const float* c1W     = (const float*)d_in[21];
    const float* c1b     = (const float*)d_in[22];
    const float* c2W     = (const float*)d_in[23];
    const float* c2b     = (const float*)d_in[24];
    const float* c3W     = (const float*)d_in[25];
    const float* c3b     = (const float*)d_in[26];
    float* out = (float*)d_out;

    // ---- workspace carve-up -------------------------------------------------
    char* p = (char*)d_ws;
    auto carve = [&](size_t bytes) -> char* {
        char* r = p;
        p += (bytes + 255) & ~(size_t)255;
        return r;
    };
    float*    h    = (float*)   carve((size_t)NN * HH * 4);
    float*    pre  = (float*)   carve((size_t)NN * HH * 4);
    float*    agg  = (float*)   carve((size_t)NN * HH * 4);
    _Float16* hf16 = (_Float16*)carve((size_t)NN * HH * 2);
    float*    deg  = (float*)   carve((size_t)NN * 4);
    float*    S1   = (float*)   carve(HH * 4);
    float*    S2   = (float*)   carve(HH * 4);
    _Float16* wEmb2 = (_Float16*)carve((size_t)4 * 8 * 512 * 2);
    _Float16* wM1   = (_Float16*)carve((size_t)LL * 9 * 8 * 512 * 2);
    _Float16* wM2   = (_Float16*)carve((size_t)LL * 4 * 8 * 512 * 2);
    _Float16* wU1   = (_Float16*)carve((size_t)LL * 9 * 8 * 512 * 2);
    _Float16* wU2   = (_Float16*)carve((size_t)LL * 4 * 8 * 512 * 2);

    // ---- degree + weight packing -------------------------------------------
    hipMemsetAsync(deg, 0, (size_t)NN * 4, stream);
    k_deg<<<(EE + 255) / 256, 256, 0, stream>>>(ei, deg, EE);
    k_pack<<<256, 256, 0, stream>>>(emb_W2, wEmb2, 1,  HH,    4);
    k_pack<<<512, 256, 0, stream>>>(m1_W,   wM1,   LL, K_MSG, 9);
    k_pack<<<512, 256, 0, stream>>>(m2_W,   wM2,   LL, HH,    4);
    k_pack<<<512, 256, 0, stream>>>(u1_W,   wU1,   LL, K_UPD, 9);
    k_pack<<<512, 256, 0, stream>>>(u2_W,   wU2,   LL, HH,    4);

    // ---- embedding ----------------------------------------------------------
    hipMemsetAsync(S1, 0, HH * 4, stream);
    hipMemsetAsync(S2, 0, HH * 4, stream);
    k_emb1<<<512, 256, 0, stream>>>(u, pos, emb_W1, emb_b1, pre, S1, S2, NN);
    k_bn_apply<<<(NN * HH + 255) / 256, 256, 0, stream>>>(pre, agg, hf16, emb_g1,
                                                          emb_be1, S1, S2, 1, NN);
    hipMemsetAsync(S1, 0, HH * 4, stream);
    hipMemsetAsync(S2, 0, HH * 4, stream);
    k_gemm64<<<(NN + 63) / 64, 128, 0, stream>>>(hf16, wEmb2, emb_b2, pre, S1, S2, NN);
    k_bn_apply<<<(NN * HH + 255) / 256, 256, 0, stream>>>(pre, h, hf16, emb_g2,
                                                          emb_be2, S1, S2, 0, NN);

    // ---- message-passing layers --------------------------------------------
    for (int i = 0; i < LL; ++i) {
        hipMemsetAsync(agg, 0, (size_t)NN * HH * 4, stream);
        k_edge<<<EE / 64, 128, 0, stream>>>(
            hf16, u, pos, ei,
            wM1 + (size_t)i * 9 * 8 * 512, m1_b + i * HH,
            wM2 + (size_t)i * 4 * 8 * 512, m2_b + i * HH,
            agg, EE);
        hipMemsetAsync(S1, 0, HH * 4, stream);
        hipMemsetAsync(S2, 0, HH * 4, stream);
        k_update<<<(NN + 63) / 64, 128, 0, stream>>>(
            hf16, h, agg, deg, pos,
            wU1 + (size_t)i * 9 * 8 * 512, u1_b + i * HH,
            wU2 + (size_t)i * 4 * 8 * 512, u2_b + i * HH,
            pre, S1, S2, NN);
        k_bn_apply<<<(NN * HH + 255) / 256, 256, 0, stream>>>(pre, h, hf16,
                                                              bn_g + i * HH,
                                                              bn_b + i * HH,
                                                              S1, S2, 0, NN);
    }

    // ---- conv head ----------------------------------------------------------
    k_head<<<(NN + 63) / 64, 64, 0, stream>>>(h, c1W, c1b, c2W, c2b, c3W, c3b,
                                              out, NN);
}
